// OptimizedASAMLayer_12120397709583
// MI455X (gfx1250) — compile-verified, matrix operand order checked by reading, not run on hardware
//
#include <hip/hip_runtime.h>
#include <hip/hip_bf16.h>
#include <math.h>

// ---------------------------------------------------------------------------
// Types for CDNA5 WMMA (wave32): v_wmma_f32_16x16x32_f16
// ---------------------------------------------------------------------------
typedef __attribute__((ext_vector_type(16))) _Float16 v16h;
typedef __attribute__((ext_vector_type(8)))  _Float16 v8h;
typedef __attribute__((ext_vector_type(4)))  _Float16 v4h;
typedef __attribute__((ext_vector_type(2)))  _Float16 v2h;
typedef __attribute__((ext_vector_type(8)))  float    v8f;

#define S_LEN 2048
#define D_DIM 512
#define H_NUM 8
#define DH_DIM 64
#define QKV_STRIDE 1536   // 3*D

#define BM 128
#define BN 128
#define BK 32

// LDS byte offset of a __shared__ object = low 32 bits of its flat address
// (flat-LDS aperture: LDS address is the low 32 bits, wave-relative — same convention
//  the compiler uses for ds_* addressing).
#define LDS_OFF(p) ((unsigned)(unsigned long long)(void*)(p))

// CDNA5 async global->LDS copy, 16B per lane, tracked by ASYNCcnt.
__device__ __forceinline__ void async_ld_b128(unsigned lds_byte_off, const void* gaddr) {
    asm volatile("global_load_async_to_lds_b128 %0, %1, off"
                 :: "v"(lds_byte_off), "v"(gaddr) : "memory");
}
__device__ __forceinline__ void wait_async0() {
    asm volatile("s_wait_asynccnt 0x0" ::: "memory");
}

// ---------------------------------------------------------------------------
// Kernel 1: LayerNorm + cast to f16 (one block per row, 256 threads)
// ---------------------------------------------------------------------------
__global__ __launch_bounds__(256) void ln_cast_kernel(
    const float* __restrict__ x, const float* __restrict__ g,
    const float* __restrict__ b, _Float16* __restrict__ xh)
{
    const int row = blockIdx.x;
    const int tid = threadIdx.x;
    const float* xr = x + (size_t)row * D_DIM;
    float v0 = xr[tid];
    float v1 = xr[tid + 256];

    __shared__ float red[8];
    __shared__ float stat[2];

    float s = v0 + v1;
    #pragma unroll
    for (int o = 16; o > 0; o >>= 1) s += __shfl_xor(s, o, 32);
    if ((tid & 31) == 0) red[tid >> 5] = s;
    __syncthreads();
    if (tid == 0) {
        float t = 0.f;
        #pragma unroll
        for (int i = 0; i < 8; ++i) t += red[i];
        stat[0] = t * (1.0f / (float)D_DIM);
    }
    __syncthreads();
    const float mu = stat[0];
    const float d0 = v0 - mu, d1 = v1 - mu;

    s = d0 * d0 + d1 * d1;
    #pragma unroll
    for (int o = 16; o > 0; o >>= 1) s += __shfl_xor(s, o, 32);
    if ((tid & 31) == 0) red[tid >> 5] = s;
    __syncthreads();
    if (tid == 0) {
        float t = 0.f;
        #pragma unroll
        for (int i = 0; i < 8; ++i) t += red[i];
        stat[1] = t * (1.0f / (float)D_DIM);
    }
    __syncthreads();
    const float rstd = rsqrtf(stat[1] + 1e-5f);

    xh[(size_t)row * D_DIM + tid]       = (_Float16)(d0 * rstd * g[tid] + b[tid]);
    xh[(size_t)row * D_DIM + tid + 256] = (_Float16)(d1 * rstd * g[tid + 256] + b[tid + 256]);
}

// ---------------------------------------------------------------------------
// Kernel 2: transpose + cast weights f32 (K x N) -> f16 (N x K)
// ---------------------------------------------------------------------------
__global__ __launch_bounds__(256) void transpose_cast_kernel(
    const float* __restrict__ w, _Float16* __restrict__ wt, int K, int N)
{
    int idx = blockIdx.x * 256 + threadIdx.x;
    if (idx < K * N) {
        int k = idx / N;
        int n = idx - k * N;
        wt[(size_t)n * K + k] = (_Float16)w[idx];
    }
}

// ---------------------------------------------------------------------------
// Kernel 3: tiled WMMA GEMM with double-buffered async global->LDS pipeline.
// C[M,N] = A[M,K](f16) * Bt[N,K](f16)^T + bias  (+gelu) (+res), f32/f16 out.
// 256 threads = 8 waves; block tile 128x128; wave tile 32x64; BK = 32.
// ---------------------------------------------------------------------------
__global__ __launch_bounds__(256) void gemm_wmma_kernel(
    const _Float16* __restrict__ A,    // M x K
    const _Float16* __restrict__ Bt,   // N x K (transposed weight)
    const float*    __restrict__ bias, // N
    const float*    __restrict__ res,  // M x N residual or nullptr
    float*          __restrict__ outf, // M x N f32 out or nullptr
    _Float16*       __restrict__ outh, // M x N f16 out or nullptr
    int M, int N, int K, int gelu_flag)
{
    __shared__ __align__(16) _Float16 lA[2][BM * BK];   // 2 x 8 KB
    __shared__ __align__(16) _Float16 lB[2][BN * BK];   // 2 x 8 KB

    const int tid  = threadIdx.x;
    const int lane = tid & 31;
    const int wave = tid >> 5;
    const int wm   = wave >> 1;   // 0..3  -> 32-row slab
    const int wn   = wave & 1;    // 0..1  -> 64-col slab
    const int bm   = blockIdx.y * BM;
    const int bn   = blockIdx.x * BN;

    const int mlo = lane & 15;
    const int grp = lane >> 4;

    v8f acc[2][4] = {};

    // issue async tile load for K-offset k0 into buffer `buf`
    auto issue_tile = [&](int k0, int buf) {
        #pragma unroll
        for (int i = 0; i < 2; ++i) {
            const int c   = tid + i * 256;       // 0..511 chunk id (16B chunks)
            const int row = c >> 2;              // 4 chunks per 32-half row
            const int seg = (c & 3) * 8;         // halves
            async_ld_b128(LDS_OFF(&lA[buf][row * BK + seg]),
                          &A [(size_t)(bm + row) * K + k0 + seg]);
            async_ld_b128(LDS_OFF(&lB[buf][row * BK + seg]),
                          &Bt[(size_t)(bn + row) * K + k0 + seg]);
        }
    };

    issue_tile(0, 0);
    wait_async0();
    __syncthreads();

    for (int k0 = 0; k0 < K; k0 += BK) {
        const int buf = (k0 >> 5) & 1;
        const bool more = (k0 + BK) < K;
        if (more) issue_tile(k0 + BK, buf ^ 1);   // overlap with compute below

        // build fragments (ISA 16-bit A-matrix 16x32 lane layout)
        v16h af[2], bf[4];
        #pragma unroll
        for (int i = 0; i < 2; ++i) {
            const int r = wm * 32 + i * 16 + mlo;
            ((v8h*)&af[i])[0] = *(const v8h*)&lA[buf][r * BK + grp * 8];
            ((v8h*)&af[i])[1] = *(const v8h*)&lA[buf][r * BK + 16 + grp * 8];
        }
        #pragma unroll
        for (int j = 0; j < 4; ++j) {
            const int c = wn * 64 + j * 16 + mlo;
            ((v8h*)&bf[j])[0] = *(const v8h*)&lB[buf][c * BK + grp * 8];
            ((v8h*)&bf[j])[1] = *(const v8h*)&lB[buf][c * BK + 16 + grp * 8];
        }

        #pragma unroll
        for (int i = 0; i < 2; ++i)
            #pragma unroll
            for (int j = 0; j < 4; ++j)
                acc[i][j] = __builtin_amdgcn_wmma_f32_16x16x32_f16(
                    false, af[i], false, bf[j], (short)0, acc[i][j], false, false);

        if (more) wait_async0();   // my next-tile loads have landed
        __syncthreads();           // everyone's have landed; buf reads done
    }

    // epilogue: bias (+gelu) (+residual), dual-precision store
    const int nlo = lane & 15;
    const int hi  = lane >> 4;
    #pragma unroll
    for (int i = 0; i < 2; ++i) {
        #pragma unroll
        for (int j = 0; j < 4; ++j) {
            const int col = bn + wn * 64 + j * 16 + nlo;
            const float bv = bias ? bias[col] : 0.0f;
            #pragma unroll
            for (int r = 0; r < 8; ++r) {
                const int row = bm + wm * 32 + i * 16 + hi * 8 + r;
                float v = acc[i][j][r] + bv;
                if (gelu_flag) v = 0.5f * v * (1.0f + erff(v * 0.70710678118654752f));
                const size_t idx = (size_t)row * N + col;
                if (res)  v += res[idx];
                if (outf) outf[idx] = v;
                if (outh) outh[idx] = (_Float16)v;
            }
        }
    }
}

// ---------------------------------------------------------------------------
// Kernel 4: local windowed attention (W=129) with LDS-staged K/V window.
// Block = (8 queries) x (1 head); 256 threads = 8 waves, one wave per query.
// Window rows are staged zero-padded, which reproduces the reference's
// zero-pad semantics exactly (padded key -> logit 0, padded V -> no contrib).
// ---------------------------------------------------------------------------
#define WROWS 136   // 8 queries + 128 window
__global__ __launch_bounds__(256) void local_attn_kernel(
    const float* __restrict__ qkv, _Float16* __restrict__ attn_h)
{
    const int tid  = threadIdx.x;
    const int lane = tid & 31;
    const int wave = tid >> 5;
    const int s_base = blockIdx.x * 8;
    const int h = blockIdx.y;

    __shared__ __align__(16) _Float16 ldsK[WROWS][DH_DIM];  // 17 KB
    __shared__ __align__(16) _Float16 ldsV[WROWS][DH_DIM];  // 17 KB
    __shared__ float attnw[8][132];

    // ---- stage K/V window (f32 -> f16), zero-fill out-of-range rows ----
    for (int c = tid; c < WROWS * (DH_DIM / 4); c += 256) {
        const int p  = c >> 4;
        const int dq = (c & 15) * 4;
        const int pos = s_base - 64 + p;
        float4 kv = make_float4(0.f, 0.f, 0.f, 0.f);
        float4 vv = make_float4(0.f, 0.f, 0.f, 0.f);
        if (pos >= 0 && pos < S_LEN) {
            const size_t base = (size_t)pos * QKV_STRIDE + h * DH_DIM + dq;
            kv = *(const float4*)&qkv[base + 512];
            vv = *(const float4*)&qkv[base + 1024];
        }
        v4h kh = { (_Float16)kv.x, (_Float16)kv.y, (_Float16)kv.z, (_Float16)kv.w };
        v4h vh = { (_Float16)vv.x, (_Float16)vv.y, (_Float16)vv.z, (_Float16)vv.w };
        *(v4h*)&ldsK[p][dq] = kh;
        *(v4h*)&ldsV[p][dq] = vh;
    }
    __syncthreads();

    // ---- per-wave query ----
    const int s = s_base + wave;
    const float* qr = qkv + (size_t)s * QKV_STRIDE + h * DH_DIM;
    float qreg[DH_DIM];
    #pragma unroll
    for (int d = 0; d < DH_DIM; ++d) qreg[d] = qr[d];

    // scores: lane covers j = lane, lane+32, ... (window W = 129)
    float sc[5];
    #pragma unroll
    for (int i = 0; i < 5; ++i) {
        const int j = lane + i * 32;
        float r = -INFINITY;                        // j >= 129: excluded
        if (j < 129) {
            const int p = wave + j;                 // staged row
            float acc = 0.f;
            const v8h* krow = (const v8h*)&ldsK[p][0];
            #pragma unroll
            for (int cc = 0; cc < 8; ++cc) {
                v8h kh = krow[cc];
                #pragma unroll
                for (int u = 0; u < 8; ++u)
                    acc = fmaf(qreg[cc * 8 + u], (float)kh[u], acc);
            }
            r = acc * 0.125f;                       // 1/sqrt(64)
        }
        sc[i] = r;
    }

    // wave-wide softmax
    float m = sc[0];
    #pragma unroll
    for (int i = 1; i < 5; ++i) m = fmaxf(m, sc[i]);
    #pragma unroll
    for (int o = 16; o > 0; o >>= 1) m = fmaxf(m, __shfl_xor(m, o, 32));
    float ssum = 0.f, e[5];
    #pragma unroll
    for (int i = 0; i < 5; ++i) {
        e[i] = (sc[i] == -INFINITY) ? 0.f : __expf(sc[i] - m);
        ssum += e[i];
    }
    #pragma unroll
    for (int o = 16; o > 0; o >>= 1) ssum += __shfl_xor(ssum, o, 32);
    const float inv = 1.0f / ssum;
    #pragma unroll
    for (int i = 0; i < 5; ++i) {
        const int j = lane + i * 32;
        if (j < 129) attnw[wave][j] = e[i] * inv;
    }
    __syncthreads();

    // weighted sum of V: each lane owns 2 of the 64 dims (zero rows are free)
    const int d0 = lane * 2;
    float a0 = 0.f, a1 = 0.f;
    for (int j = 0; j < 129; ++j) {
        const float w = attnw[wave][j];
        v2h vp = *(const v2h*)&ldsV[wave + j][d0];
        a0 = fmaf(w, (float)vp[0], a0);
        a1 = fmaf(w, (float)vp[1], a1);
    }
    attn_h[(size_t)s * D_DIM + h * DH_DIM + d0]     = (_Float16)a0;
    attn_h[(size_t)s * D_DIM + h * DH_DIM + d0 + 1] = (_Float16)a1;
}

// ---------------------------------------------------------------------------
// Host-side orchestration
// ---------------------------------------------------------------------------
extern "C" void kernel_launch(void* const* d_in, const int* in_sizes, int n_in,
                              void* d_out, int out_size, void* d_ws, size_t ws_size,
                              hipStream_t stream) {
    const float* x     = (const float*)d_in[0];
    const float* w_qkv = (const float*)d_in[1];
    const float* b_qkv = (const float*)d_in[2];
    const float* w_out = (const float*)d_in[3];
    const float* b_out = (const float*)d_in[4];
    const float* ln_g  = (const float*)d_in[5];
    const float* ln_b  = (const float*)d_in[6];
    const float* w1    = (const float*)d_in[7];
    const float* b1    = (const float*)d_in[8];
    const float* w2    = (const float*)d_in[9];
    const float* b2    = (const float*)d_in[10];
    float* out = (float*)d_out;

    // workspace carve-up
    char* w = (char*)d_ws;
    _Float16* xn_h   = (_Float16*)w; w += (size_t)S_LEN * D_DIM * 2;       // 2 MB
    _Float16* wtqkv  = (_Float16*)w; w += (size_t)QKV_STRIDE * D_DIM * 2;  // 1.5 MB
    _Float16* wtout  = (_Float16*)w; w += (size_t)D_DIM * D_DIM * 2;       // 0.5 MB
    _Float16* wt1    = (_Float16*)w; w += (size_t)(4 * D_DIM) * D_DIM * 2; // 2 MB
    _Float16* wt2    = (_Float16*)w; w += (size_t)D_DIM * (4 * D_DIM) * 2; // 2 MB
    float*    qkv_f  = (float*)w;    w += (size_t)S_LEN * QKV_STRIDE * 4;  // 12 MB
    float*    out_f  = (float*)w;    w += (size_t)S_LEN * D_DIM * 4;       // 4 MB
    _Float16* attn_h = (_Float16*)w; w += (size_t)S_LEN * D_DIM * 2;       // 2 MB
    _Float16* out_h  = (_Float16*)w; w += (size_t)S_LEN * D_DIM * 2;       // 2 MB
    _Float16* h1_h   = (_Float16*)w; w += (size_t)S_LEN * (4 * D_DIM) * 2; // 8 MB

    // 1. LayerNorm + f16 cast
    ln_cast_kernel<<<S_LEN, 256, 0, stream>>>(x, ln_g, ln_b, xn_h);

    // 2. weight transpose+cast (f32 KxN -> f16 NxK)
    {
        int n;
        n = D_DIM * QKV_STRIDE;
        transpose_cast_kernel<<<(n + 255) / 256, 256, 0, stream>>>(w_qkv, wtqkv, D_DIM, QKV_STRIDE);
        n = D_DIM * D_DIM;
        transpose_cast_kernel<<<(n + 255) / 256, 256, 0, stream>>>(w_out, wtout, D_DIM, D_DIM);
        n = D_DIM * 4 * D_DIM;
        transpose_cast_kernel<<<(n + 255) / 256, 256, 0, stream>>>(w1, wt1, D_DIM, 4 * D_DIM);
        transpose_cast_kernel<<<(n + 255) / 256, 256, 0, stream>>>(w2, wt2, 4 * D_DIM, D_DIM);
    }

    // 3. QKV projection: qkv = xn @ w_qkv + b_qkv  (2048 x 1536, K=512)
    gemm_wmma_kernel<<<dim3(QKV_STRIDE / BN, S_LEN / BM), 256, 0, stream>>>(
        xn_h, wtqkv, b_qkv, nullptr, qkv_f, nullptr, S_LEN, QKV_STRIDE, D_DIM, 0);

    // 4. local attention -> attn_h (f16)
    local_attn_kernel<<<dim3(S_LEN / 8, H_NUM), 256, 0, stream>>>(qkv_f, attn_h);

    // 5. out-projection + residual: out = x + attn @ w_out + b_out (f32 + f16)
    gemm_wmma_kernel<<<dim3(D_DIM / BN, S_LEN / BM), 256, 0, stream>>>(
        attn_h, wtout, b_out, x, out_f, out_h, S_LEN, D_DIM, D_DIM, 0);

    // 6. FFN up + exact GELU: h1 = gelu(out @ w1 + b1)  (2048 x 2048, K=512)
    gemm_wmma_kernel<<<dim3(4 * D_DIM / BN, S_LEN / BM), 256, 0, stream>>>(
        out_h, wt1, b1, nullptr, nullptr, h1_h, S_LEN, 4 * D_DIM, D_DIM, 1);

    // 7. FFN down + residual: final = out + h1 @ w2 + b2 (2048 x 512, K=2048)
    gemm_wmma_kernel<<<dim3(D_DIM / BN, S_LEN / BM), 256, 0, stream>>>(
        h1_h, wt2, b2, out_f, out, nullptr, S_LEN, D_DIM, 4 * D_DIM, 0);
}
